// EdgeBlock_21509196219221
// MI455X (gfx1250) — compile-verified, matrix operand order checked by reading, not run on hardware
//
#include <hip/hip_runtime.h>

// ---------------------------------------------------------------------------
// EdgeBlock MLP for MI455X (gfx1250, wave32, WMMA).
//   x = [edge_feat | node_feat[senders] | node_feat[receivers]]  (E x 384) f32
//   h = relu(x @ W1 + b1)   (E x 256)
//   y = h @ W2 + b2         (E x 128) f32
//
// bf16 WMMA 16x16x32, f32 accumulate. Weights pre-packed (prep kernel) into
// per-lane B-fragment layout in d_ws. One 256-thread block (8 waves) handles
// 128 edges. Waves split the N dimension so each packed B fragment is fetched
// from L2 once (GEMM1) / twice (GEMM2) per block. Each A-fragment LDS load
// feeds TWO wmmas (dual-N accumulators) to stay under LDS bandwidth.
// LDS tiles use a 16-byte-chunk XOR swizzle (chunk ^ (row & 15)) so the
// 16-rows-per-lane A-fragment ds_load_b128s are bank-conflict free with no
// padding: xs 96 KB + hs 64 KB = 160 KB -> 2 blocks per 320 KB WGP.
// ---------------------------------------------------------------------------

typedef __bf16 bf16_t;
typedef __attribute__((ext_vector_type(16))) __bf16 v16bf;
typedef __attribute__((ext_vector_type(8)))  __bf16 v8bf;
typedef __attribute__((ext_vector_type(8)))  float  v8f;
typedef __attribute__((ext_vector_type(4)))  float  v4f;

#define D_FEAT 128
#define HIDDEN 256
#define XDIM   384        // 3 * D_FEAT
#define M_TILE 128        // edges per block
#define MT     8          // 128 / 16 M-tiles per block
#define KT1    12         // 384 / 32
#define NT1    16         // 256 / 16
#define KT2    8          // 256 / 32
#define NT2    8          // 128 / 16
#define TILE_ELEMS 512    // 32 lanes * 16 bf16 per packed B fragment
#define XCH    48         // 16-byte chunks per xs row (384 bf16)

static __device__ __forceinline__ bf16_t f2bf(float f) {
    // round-to-nearest-even f32 -> bf16
    unsigned u = __float_as_uint(f);
    u = u + 0x7FFFu + ((u >> 16) & 1u);
    unsigned short s = (unsigned short)(u >> 16);
    union { unsigned short s; bf16_t b; } cv;
    cv.s = s;
    return cv.b;
}

// XOR swizzle of a 16B chunk index by the row's low 4 bits: bank-conflict-free
// column reads across 16 consecutive rows.
static __device__ __forceinline__ int swz(int c, int row) {
    return (c & ~15) | ((c & 15) ^ (row & 15));
}

// K index inside a 32-deep fragment for element e (0..15) given lane half
// (CDNA5 16-bit A-matrix 16x32 layout, ISA 7.12.2).
static __device__ __forceinline__ int kidx(int e, int half) {
    return (e < 8) ? (half * 8 + e) : (16 + half * 8 + (e - 8));
}

// ---------------------------------------------------------------------------
// Prep: pack W1 (384x256) and W2 (256x128) f32 row-major -> bf16 B fragments.
// Fragment (nt,kt): lane l owns 16 contiguous bf16 = column n = nt*16+(l&15),
// K values kt*32 + kidx(e, l>>4)  (B column-major per lane, mirroring A).
// ---------------------------------------------------------------------------
__global__ void pack_weights_kernel(const float* __restrict__ W1,
                                    const float* __restrict__ W2,
                                    bf16_t* __restrict__ pW1,
                                    bf16_t* __restrict__ pW2) {
    int tid  = blockIdx.x * blockDim.x + threadIdx.x;
    int lane = tid & 31;
    int tile = tid >> 5;
    int half = lane >> 4;
    int ncol = lane & 15;
    const int nW1 = NT1 * KT1;          // 192 tiles
    if (tile < nW1) {
        int nt = tile / KT1, kt = tile % KT1;
        bf16_t* dst = pW1 + (size_t)tile * TILE_ELEMS + lane * 16;
        int n = nt * 16 + ncol;
#pragma unroll
        for (int e = 0; e < 16; ++e) {
            int k = kt * 32 + kidx(e, half);
            dst[e] = f2bf(W1[(size_t)k * HIDDEN + n]);
        }
    } else if (tile < nW1 + NT2 * KT2) { // 64 tiles
        int t2 = tile - nW1;
        int nt = t2 / KT2, kt = t2 % KT2;
        bf16_t* dst = pW2 + (size_t)t2 * TILE_ELEMS + lane * 16;
        int n = nt * 16 + ncol;
#pragma unroll
        for (int e = 0; e < 16; ++e) {
            int k = kt * 32 + kidx(e, half);
            dst[e] = f2bf(W2[(size_t)k * D_FEAT + n]);
        }
    }
}

// ---------------------------------------------------------------------------
// Main fused kernel: gather + concat + MLP via WMMA.
// ---------------------------------------------------------------------------
__global__ __launch_bounds__(256, 1)
void edge_mlp_kernel(const float* __restrict__ edge_feat,
                     const float* __restrict__ node_feat,
                     const int*   __restrict__ senders,
                     const int*   __restrict__ receivers,
                     const bf16_t* __restrict__ pW1,
                     const bf16_t* __restrict__ pW2,
                     const float* __restrict__ b1,
                     const float* __restrict__ b2,
                     float* __restrict__ out,
                     int E) {
    __shared__ __attribute__((aligned(16))) bf16_t xs[M_TILE * XDIM];   // 96 KB
    __shared__ __attribute__((aligned(16))) bf16_t hs[M_TILE * HIDDEN]; // 64 KB

    const int tid  = threadIdx.x;
    const int lane = tid & 31;
    const int wave = tid >> 5;          // 0..7
    const int half = lane >> 4;
    const int rc   = lane & 15;         // row (A) / column (B,C,D) within tile
    const int e0   = blockIdx.x * M_TILE;
    const bool full = (e0 + M_TILE) <= E;   // whole tile in-range?

    // --- Stage bf16 x-tile: 128 rows x 48 swizzled 16B chunks ---------------
    for (int idx = tid; idx < M_TILE * XCH; idx += 256) {
        int row = idx / XCH;
        int c   = idx % XCH;            // 16B chunk in row = 8 floats of input
        int ge  = e0 + row;
        if (ge >= E) ge = E - 1;        // clamp (E is a multiple of 128 anyway)
        int col = c * 8;
        const float* src;
        if (col < D_FEAT)
            src = edge_feat + (size_t)ge * D_FEAT + col;
        else if (col < 2 * D_FEAT)
            src = node_feat + (size_t)senders[ge]   * D_FEAT + (col - D_FEAT);
        else
            src = node_feat + (size_t)receivers[ge] * D_FEAT + (col - 2 * D_FEAT);
        v4f v0 = *(const v4f*)src;
        v4f v1 = *(const v4f*)(src + 4);
        v8bf o;
        o[0] = f2bf(v0.x); o[1] = f2bf(v0.y); o[2] = f2bf(v0.z); o[3] = f2bf(v0.w);
        o[4] = f2bf(v1.x); o[5] = f2bf(v1.y); o[6] = f2bf(v1.z); o[7] = f2bf(v1.w);
        *(v8bf*)(xs + (size_t)row * XDIM + swz(c, row) * 8) = o;
    }
    __syncthreads();

    // --- GEMM1: wave w owns N-tiles {2w, 2w+1}, sweeps all 8 M-tiles.
    //     Both N accumulators live simultaneously so each A LDS load feeds
    //     two wmmas.
    {
        v8f acc[2][MT] = {};
        const bf16_t* bp0 = pW1 + (size_t)((2 * wave)     * KT1) * TILE_ELEMS + lane * 16;
        const bf16_t* bp1 = pW1 + (size_t)((2 * wave + 1) * KT1) * TILE_ELEMS + lane * 16;
        for (int kt = 0; kt < KT1; ++kt) {
            v16bf b0 = *(const v16bf*)(bp0 + (size_t)kt * TILE_ELEMS);
            v16bf b1 = *(const v16bf*)(bp1 + (size_t)kt * TILE_ELEMS);
            const int c0 = kt * 4 + half;       // lo 16B chunk (K k..k+7)
            const int c1 = c0 + 2;              // hi 16B chunk (K k+16..k+23)
#pragma unroll
            for (int mt = 0; mt < MT; ++mt) {
                int row = mt * 16 + rc;
                const bf16_t* rb = xs + (size_t)row * XDIM;
                v8bf lo = *(const v8bf*)(rb + swz(c0, row) * 8);
                v8bf hi = *(const v8bf*)(rb + swz(c1, row) * 8);
                v16bf a;
#pragma unroll
                for (int q = 0; q < 8; ++q) { a[q] = lo[q]; a[q + 8] = hi[q]; }
                acc[0][mt] = __builtin_amdgcn_wmma_f32_16x16x32_bf16(
                    false, a, false, b0, (short)0, acc[0][mt], false, false);
                acc[1][mt] = __builtin_amdgcn_wmma_f32_16x16x32_bf16(
                    false, a, false, b1, (short)0, acc[1][mt], false, false);
            }
        }
        // bias + ReLU + bf16 store to hs (swizzled)
#pragma unroll
        for (int i = 0; i < 2; ++i) {
            const int n = (2 * wave + i) * 16 + rc;
            const float bias = b1[n];
#pragma unroll
            for (int mt = 0; mt < MT; ++mt) {
#pragma unroll
                for (int j = 0; j < 8; ++j) {
                    float h = acc[i][mt][j] + bias;
                    h = h > 0.0f ? h : 0.0f;
                    int row = mt * 16 + j + 8 * half;   // D: lane=col, vgpr=row
                    hs[(size_t)row * HIDDEN + swz(n >> 3, row) * 8 + (n & 7)] = f2bf(h);
                }
            }
        }
    }
    __syncthreads();   // h is produced across waves, consumed across waves

    // --- GEMM2: wave w owns N-tile pair {2(w&3), 2(w&3)+1} over M-tiles
    //     (w>>2)*4 .. +4. Again each A load feeds two wmmas.
    {
        const int mbase = (wave >> 2) * 4;      // 0 or 4
        const int np    = (wave & 3) * 2;       // 0,2,4,6
        v8f acc[2][4] = {};
        const bf16_t* bq0 = pW2 + (size_t)((np)     * KT2) * TILE_ELEMS + lane * 16;
        const bf16_t* bq1 = pW2 + (size_t)((np + 1) * KT2) * TILE_ELEMS + lane * 16;
        for (int kt = 0; kt < KT2; ++kt) {
            v16bf b0 = *(const v16bf*)(bq0 + (size_t)kt * TILE_ELEMS);
            v16bf b1 = *(const v16bf*)(bq1 + (size_t)kt * TILE_ELEMS);
            const int c0 = kt * 4 + half;
            const int c1 = c0 + 2;
#pragma unroll
            for (int m = 0; m < 4; ++m) {
                int row = (mbase + m) * 16 + rc;
                const bf16_t* rb = hs + (size_t)row * HIDDEN;
                v8bf lo = *(const v8bf*)(rb + swz(c0, row) * 8);
                v8bf hi = *(const v8bf*)(rb + swz(c1, row) * 8);
                v16bf a;
#pragma unroll
                for (int q = 0; q < 8; ++q) { a[q] = lo[q]; a[q + 8] = hi[q]; }
                acc[0][m] = __builtin_amdgcn_wmma_f32_16x16x32_bf16(
                    false, a, false, b0, (short)0, acc[0][m], false, false);
                acc[1][m] = __builtin_amdgcn_wmma_f32_16x16x32_bf16(
                    false, a, false, b1, (short)0, acc[1][m], false, false);
            }
        }
#pragma unroll
        for (int i = 0; i < 2; ++i) {
            const int n = (np + i) * 16 + rc;
            const float bias = b2[n];
            if (full) {
#pragma unroll
                for (int m = 0; m < 4; ++m) {
#pragma unroll
                    for (int j = 0; j < 8; ++j) {
                        int row = (mbase + m) * 16 + j + 8 * half;
                        out[(size_t)(e0 + row) * D_FEAT + n] = acc[i][m][j] + bias;
                    }
                }
            } else {
#pragma unroll
                for (int m = 0; m < 4; ++m) {
#pragma unroll
                    for (int j = 0; j < 8; ++j) {
                        int row = (mbase + m) * 16 + j + 8 * half;
                        int ge  = e0 + row;
                        if (ge < E)
                            out[(size_t)ge * D_FEAT + n] = acc[i][m][j] + bias;
                    }
                }
            }
        }
    }
}

// ---------------------------------------------------------------------------
// Host-side launch. Inputs (setup_inputs order):
//   0 edge_feat f32 [E,128]   1 node_feat f32 [10000,128]
//   2 senders   i32 [E]       3 receivers i32 [E]
//   4 W1 f32 [384,256]        5 b1 f32 [256]
//   6 W2 f32 [256,128]        7 b2 f32 [128]
// Output: f32 [E,128].  d_ws holds packed bf16 weights (256 KB needed).
// ---------------------------------------------------------------------------
extern "C" void kernel_launch(void* const* d_in, const int* in_sizes, int n_in,
                              void* d_out, int out_size, void* d_ws, size_t ws_size,
                              hipStream_t stream) {
    const float* edge_feat = (const float*)d_in[0];
    const float* node_feat = (const float*)d_in[1];
    const int*   senders   = (const int*)d_in[2];
    const int*   receivers = (const int*)d_in[3];
    const float* W1        = (const float*)d_in[4];
    const float* b1        = (const float*)d_in[5];
    const float* W2        = (const float*)d_in[6];
    const float* b2        = (const float*)d_in[7];
    float*       out       = (float*)d_out;

    const int E = in_sizes[2];          // number of edges (640000)

    bf16_t* pW1 = (bf16_t*)d_ws;                       // 384*256 bf16 = 192 KB
    bf16_t* pW2 = pW1 + (size_t)XDIM * HIDDEN;         // 256*128 bf16 =  64 KB

    const int pack_threads = (NT1 * KT1 + NT2 * KT2) * 32;   // 8192
    pack_weights_kernel<<<(pack_threads + 255) / 256, 256, 0, stream>>>(
        W1, W2, pW1, pW2);

    const int nblocks = (E + M_TILE - 1) / M_TILE;     // 5000
    edge_mlp_kernel<<<nblocks, 256, 0, stream>>>(
        edge_feat, node_feat, senders, receivers, pW1, pW2, b1, b2, out, E);
}
